// MultiTaskModel_39496519254702
// MI455X (gfx1250) — compile-verified, matrix operand order checked by reading, not run on hardware
//
#include <hip/hip_runtime.h>
#include <hip/hip_bf16.h>

typedef __attribute__((ext_vector_type(16))) __bf16 v16bf;
typedef __attribute__((ext_vector_type(8)))  float  v8f;

#define NNODES 50000
#define NEDGES 600000
#define NGRAPH 64
#define SEQD   384
#define HID    128

// ---------------------------------------------------------------------------
// small utility kernels
// ---------------------------------------------------------------------------
__global__ void fill_kernel(float* __restrict__ p, float v, int n) {
  int i = blockIdx.x * blockDim.x + threadIdx.x;
  if (i < n) p[i] = v;
}

__global__ void accum_deg_kernel(const int* __restrict__ dst, float* __restrict__ deg, int e) {
  int i = blockIdx.x * blockDim.x + threadIdx.x;
  if (i < e) unsafeAtomicAdd(&deg[dst[i]], 1.0f);
}

__global__ void rsqrt_kernel(float* __restrict__ p, int n) {
  int i = blockIdx.x * blockDim.x + threadIdx.x;
  if (i < n) p[i] = rsqrtf(p[i]);
}

// Wt[c][k] (bf16) = W[k][c] (f32); W is [K,128] row-major
__global__ void wtrans_kernel(const float* __restrict__ W, __bf16* __restrict__ Wt, int K) {
  int i = blockIdx.x * blockDim.x + threadIdx.x;
  if (i >= K * HID) return;
  int c = i / K, k = i - c * K;
  Wt[i] = (__bf16)W[k * HID + c];
}

// out[i][c] = h[i][c] * dinv[i]^2   (self-loop term; also initializes buffer)
__global__ void self_init_kernel(const float* __restrict__ h, const float* __restrict__ dinv,
                                 float* __restrict__ out, int total) {
  int i = blockIdx.x * blockDim.x + threadIdx.x;
  if (i < total) {
    float d = dinv[i >> 7];
    out[i] = h[i] * d * d;
  }
}

// one wave per edge, one float4 of channels per lane
__global__ __launch_bounds__(256)
void edge_scatter_kernel(const float* __restrict__ h, const int* __restrict__ src,
                         const int* __restrict__ dst, const float* __restrict__ dinv,
                         float* __restrict__ out, int nedges) {
  long long gid = (long long)blockIdx.x * blockDim.x + threadIdx.x;
  int e = (int)(gid >> 5);
  int g = (int)(gid & 31);
  if (e >= nedges) return;
  int s = src[e], d = dst[e];
  float nrm = dinv[s] * dinv[d];
  float4 v = *(const float4*)(h + (size_t)s * HID + g * 4);
  float* op = out + (size_t)d * HID + g * 4;
  unsafeAtomicAdd(op + 0, v.x * nrm);
  unsafeAtomicAdd(op + 1, v.y * nrm);
  unsafeAtomicAdd(op + 2, v.z * nrm);
  unsafeAtomicAdd(op + 3, v.w * nrm);
}

__global__ void bias_relu_kernel(float* __restrict__ x, const float* __restrict__ b, int total) {
  int i = blockIdx.x * blockDim.x + threadIdx.x;
  if (i < total) x[i] = fmaxf(x[i] + b[i & (HID - 1)], 0.0f);
}

// relu(x + b) written directly as bf16 (feeds next WMMA GEMM's A operand)
__global__ void bias_relu_bf16_kernel(const float* __restrict__ x, const float* __restrict__ b,
                                      __bf16* __restrict__ y, int total) {
  int i = blockIdx.x * blockDim.x + threadIdx.x;
  if (i < total) y[i] = (__bf16)fmaxf(x[i] + b[i & (HID - 1)], 0.0f);
}

// ---------------------------------------------------------------------------
// WMMA GEMM: C[M,128] = A[M,K] @ W[K,128], bf16 WMMA, f32 accumulate.
// Requires M % 16 == 0 and K % 32 == 0 (true for all uses) -> no bounds
// checks in the hot loop, EXEC stays all-ones.
// B operand comes from Wt = bf16 transposed W: a 32x16 B fragment is two
// contiguous b128 loads per lane.  All 8 B fragments are loaded into
// distinct registers before the 8 WMMAs so the loads issue as one clause
// and latency overlaps the matrix ops.
// 8 waves per block; each wave owns one 16x128 output strip.
// ---------------------------------------------------------------------------
template <bool ABF16>
__global__ __launch_bounds__(256)
void gemm_wmma_n128_kernel(const void* __restrict__ Ap, const __bf16* __restrict__ Wt,
                           float* __restrict__ C, int M, int K) {
  const int lane  = threadIdx.x & 31;
  const int wave  = threadIdx.x >> 5;
  const int strip = blockIdx.x * 8 + wave;      // 16-row strip index
  if (strip * 16 >= M) return;                  // uniform per wave
  const int half = lane >> 4;                   // 0 | 1
  const int lm   = lane & 15;
  const int row  = strip * 16 + lm;

  v8f zero = {};
  v8f acc[8];
#pragma unroll
  for (int t = 0; t < 8; ++t) acc[t] = zero;

  for (int k0 = 0; k0 < K; k0 += 32) {
    // --- A fragment: lane = row `row`; element e -> K = 8*half+e (e<8),
    //     16+8*half+(e-8) (e>=8): two contiguous 8-element chunks.
    v16bf a;
    if (ABF16) {
      const __bf16* A = (const __bf16*)Ap;
      const __bf16* ap = A + (size_t)row * K + k0 + 8 * half;
      union { v16bf v; uint4 u[2]; } au;
      au.u[0] = *(const uint4*)(ap);
      au.u[1] = *(const uint4*)(ap + 16);
      a = au.v;
    } else {
      const float* A = (const float*)Ap;
#pragma unroll
      for (int v = 0; v < 8; ++v) {
        int kb = k0 + ((v < 4) ? (8 * half + 2 * v) : (16 + 8 * half + 2 * (v - 4)));
        float2 p = *(const float2*)(A + (size_t)row * K + kb);
        a[2 * v]     = (__bf16)p.x;
        a[2 * v + 1] = (__bf16)p.y;
      }
    }
    // --- load all 8 B fragments first (distinct registers, one big clause)
    v16bf bfrag[8];
#pragma unroll
    for (int t = 0; t < 8; ++t) {
      const __bf16* wp = Wt + (size_t)(t * 16 + lm) * K + k0 + 8 * half;
      union { v16bf v; uint4 u[2]; } bu;
      bu.u[0] = *(const uint4*)(wp);
      bu.u[1] = *(const uint4*)(wp + 16);
      bfrag[t] = bu.v;
    }
    // --- 8 WMMAs consume the fragments
#pragma unroll
    for (int t = 0; t < 8; ++t) {
      acc[t] = __builtin_amdgcn_wmma_f32_16x16x32_bf16(
          false, a, false, bfrag[t], (short)0, acc[t], false, false);
    }
  }

  // C/D layout: VGPR r -> (m = r + 8*half, n = lane & 15)
#pragma unroll
  for (int t = 0; t < 8; ++t) {
    int n = t * 16 + lm;
#pragma unroll
    for (int r = 0; r < 8; ++r) {
      int m = strip * 16 + r + 8 * half;
      C[(size_t)m * HID + n] = acc[t][r];
    }
  }
}

// ---------------------------------------------------------------------------
// pooling + head
// ---------------------------------------------------------------------------
__global__ void pool_accum_kernel(const float* __restrict__ g, const int* __restrict__ batch,
                                  float* __restrict__ sums, int nnodes) {
  long long gid = (long long)blockIdx.x * blockDim.x + threadIdx.x;
  int node = (int)(gid >> 5);
  int grp  = (int)(gid & 31);
  if (node >= nnodes) return;
  int b = batch[node];
  float4 v = *(const float4*)(g + (size_t)node * HID + grp * 4);
  float* sp = sums + (size_t)b * HID + grp * 4;
  unsafeAtomicAdd(sp + 0, v.x);
  unsafeAtomicAdd(sp + 1, v.y);
  unsafeAtomicAdd(sp + 2, v.z);
  unsafeAtomicAdd(sp + 3, v.w);
}

__global__ void count_accum_kernel(const int* __restrict__ batch, float* __restrict__ cnts, int n) {
  int i = blockIdx.x * blockDim.x + threadIdx.x;
  if (i < n) unsafeAtomicAdd(&cnts[batch[i]], 1.0f);
}

__global__ void finalize_pool_kernel(float* __restrict__ sums, const float* __restrict__ cnts) {
  int i = blockIdx.x * blockDim.x + threadIdx.x;
  if (i < NGRAPH * HID) sums[i] /= fmaxf(cnts[i >> 7], 1.0f);
}

// fin (bf16) = concat([seq_emb, pooled]) — written directly in WMMA A format
__global__ void concat_bf16_kernel(const float* __restrict__ seq, const float* __restrict__ pooled,
                                   __bf16* __restrict__ fin) {
  int i = blockIdx.x * blockDim.x + threadIdx.x;
  if (i >= NGRAPH * (SEQD + HID)) return;
  int row = i / (SEQD + HID);
  int c   = i - row * (SEQD + HID);
  float v = (c < SEQD) ? seq[row * SEQD + c] : pooled[row * HID + (c - SEQD)];
  fin[i] = (__bf16)v;
}

__global__ void heads_kernel(const float* __restrict__ fused,
                             const float* __restrict__ Wtp, const float* __restrict__ btp,
                             const float* __restrict__ Wp,  const float* __restrict__ bp,
                             float* __restrict__ out) {
  int t = blockIdx.x * blockDim.x + threadIdx.x;
  if (t >= NGRAPH * 3) return;
  int row = t / 3, o = t - row * 3;
  const float* f = fused + row * HID;
  float s = 0.0f;
  if (o < 2) {
    for (int k = 0; k < HID; ++k) s += f[k] * Wtp[k * 2 + o];
    out[row * 2 + o] = s + btp[o];
  } else {
    for (int k = 0; k < HID; ++k) s += f[k] * Wp[k];
    out[NGRAPH * 2 + row] = s + bp[0];
  }
}

// ---------------------------------------------------------------------------
// launch
// ---------------------------------------------------------------------------
extern "C" void kernel_launch(void* const* d_in, const int* in_sizes, int n_in,
                              void* d_out, int out_size, void* d_ws, size_t ws_size,
                              hipStream_t stream) {
  const float* seq   = (const float*)d_in[0];
  const float* nodeX = (const float*)d_in[1];
  const int*   eidx  = (const int*)d_in[2];
  const int*   batch = (const int*)d_in[3];
  const float* W1  = (const float*)d_in[4];
  const float* b1  = (const float*)d_in[5];
  const float* W2  = (const float*)d_in[6];
  const float* b2  = (const float*)d_in[7];
  const float* Wf  = (const float*)d_in[8];
  const float* bf  = (const float*)d_in[9];
  const float* Wtp = (const float*)d_in[10];
  const float* btp = (const float*)d_in[11];
  const float* Wp  = (const float*)d_in[12];
  const float* bp  = (const float*)d_in[13];
  const int* srcE = eidx;
  const int* dstE = eidx + NEDGES;

  // f32 workspace region
  float* ws    = (float*)d_ws;
  float* dinv  = ws;                                  // 50000
  float* bufA  = dinv + NNODES;                       // 6.4M
  float* bufB  = bufA + (size_t)NNODES * HID;         // 6.4M
  float* sums  = bufB + (size_t)NNODES * HID;         // 8192
  float* cnts  = sums + NGRAPH * HID;                 // 64
  float* fused = cnts + NGRAPH;                       // 8192
  // bf16 workspace region (16B-aligned sub-offsets)
  __bf16* bws  = (__bf16*)(fused + NGRAPH * HID);
  __bf16* g1b  = bws;                                 // 50000*128
  __bf16* W1t  = g1b + (size_t)NNODES * HID;          // 128*384
  __bf16* W2t  = W1t + HID * SEQD;                    // 128*128
  __bf16* Wft  = W2t + HID * HID;                     // 128*512
  __bf16* finb = Wft + HID * (SEQD + HID);            // 64*512

  const int T = 256;
  const int totNH = NNODES * HID;
  const int gemmBlocksFull = (NNODES / 16 + 7) / 8;   // 8 strips (waves) per block

  // weight transpose+convert (tiny, once per call)
  wtrans_kernel<<<(SEQD * HID + T - 1) / T, T, 0, stream>>>(W1, W1t, SEQD);
  wtrans_kernel<<<(HID * HID + T - 1) / T, T, 0, stream>>>(W2, W2t, HID);
  wtrans_kernel<<<((SEQD + HID) * HID + T - 1) / T, T, 0, stream>>>(Wf, Wft, SEQD + HID);

  // degree with self-loops -> dinv
  fill_kernel<<<(NNODES + T - 1) / T, T, 0, stream>>>(dinv, 1.0f, NNODES);
  accum_deg_kernel<<<(NEDGES + T - 1) / T, T, 0, stream>>>(dstE, dinv, NEDGES);
  rsqrt_kernel<<<(NNODES + T - 1) / T, T, 0, stream>>>(dinv, NNODES);

  // conv1: h1 = X @ W1 ; agg = A_hat h1 ; g1 = relu(agg + b1) (bf16)
  gemm_wmma_n128_kernel<false><<<gemmBlocksFull, 256, 0, stream>>>(nodeX, W1t, bufA, NNODES, SEQD);
  self_init_kernel<<<(totNH + T - 1) / T, T, 0, stream>>>(bufA, dinv, bufB, totNH);
  edge_scatter_kernel<<<(int)(((long long)NEDGES * 32 + T - 1) / T), T, 0, stream>>>(
      bufA, srcE, dstE, dinv, bufB, NEDGES);
  bias_relu_bf16_kernel<<<(totNH + T - 1) / T, T, 0, stream>>>(bufB, b1, g1b, totNH);

  // conv2: h2 = g1 @ W2 ; agg ; g2 = relu(agg + b2) (f32, feeds pooling)
  gemm_wmma_n128_kernel<true><<<gemmBlocksFull, 256, 0, stream>>>(g1b, W2t, bufA, NNODES, HID);
  self_init_kernel<<<(totNH + T - 1) / T, T, 0, stream>>>(bufA, dinv, bufB, totNH);
  edge_scatter_kernel<<<(int)(((long long)NEDGES * 32 + T - 1) / T), T, 0, stream>>>(
      bufA, srcE, dstE, dinv, bufB, NEDGES);
  bias_relu_kernel<<<(totNH + T - 1) / T, T, 0, stream>>>(bufB, b2, totNH);

  // global mean pool
  fill_kernel<<<(NGRAPH * HID + NGRAPH + T - 1) / T, T, 0, stream>>>(
      sums, 0.0f, NGRAPH * HID + NGRAPH);
  pool_accum_kernel<<<(int)(((long long)NNODES * 32 + T - 1) / T), T, 0, stream>>>(
      bufB, batch, sums, NNODES);
  count_accum_kernel<<<(NNODES + T - 1) / T, T, 0, stream>>>(batch, cnts, NNODES);
  finalize_pool_kernel<<<(NGRAPH * HID + T - 1) / T, T, 0, stream>>>(sums, cnts);

  // head: fused = relu([seq, pooled] @ Wf + bf) ; tiny output heads
  concat_bf16_kernel<<<(NGRAPH * (SEQD + HID) + T - 1) / T, T, 0, stream>>>(seq, sums, finb);
  gemm_wmma_n128_kernel<true><<<1, 256, 0, stream>>>(finb, Wft, fused, NGRAPH, SEQD + HID);
  bias_relu_kernel<<<(NGRAPH * HID + T - 1) / T, T, 0, stream>>>(fused, bf, NGRAPH * HID);
  heads_kernel<<<1, NGRAPH * 3, 0, stream>>>(fused, Wtp, btp, Wp, bp, (float*)d_out);
}
